// LinearAttention_4973572128840
// MI455X (gfx1250) — compile-verified, matrix operand order checked by reading, not run on hardware
//
#include <hip/hip_runtime.h>
#include <cstdint>

// ---------------------------------------------------------------------------
// Types for CDNA5 WMMA
// ---------------------------------------------------------------------------
typedef __attribute__((ext_vector_type(8)))  _Float16 h8;
typedef __attribute__((ext_vector_type(16))) _Float16 h16;
typedef __attribute__((ext_vector_type(8)))  __bf16   y8;
typedef __attribute__((ext_vector_type(16))) __bf16   y16;
typedef __attribute__((ext_vector_type(8)))  float    f8;
typedef unsigned short u16s;

#define DEV __device__ __forceinline__

// Problem constants
static constexpr int BATCH = 4;
static constexpr int SEQ   = 16384;
static constexpr int DIM   = 512;
static constexpr int HEADS = 8;
static constexpr int DH    = 64;
static constexpr int ROWS  = BATCH * SEQ;       // 65536
static constexpr int QKVF  = 3 * DIM;           // 1536
static constexpr int BH    = BATCH * HEADS;     // 32

// ---------------------------------------------------------------------------
// Helpers
// ---------------------------------------------------------------------------
DEV u16s f32_to_bf16(float f) {
    unsigned int u = __float_as_uint(f);
    unsigned int r = u + 0x7FFFu + ((u >> 16) & 1u);   // round-to-nearest-even
    return (u16s)(r >> 16);
}

DEV f8 f8zero() {
    f8 z;
#pragma unroll
    for (int i = 0; i < 8; ++i) z[i] = 0.0f;
    return z;
}

// WMMA wrappers: D = A(16x32) * B(32x16) + C, f32 accumulate
DEV f8 wmma32(h16 a, h16 b, f8 c) {
    return __builtin_amdgcn_wmma_f32_16x16x32_f16(false, a, false, b, (short)0, c, false, false);
}
DEV f8 wmma32(y16 a, y16 b, f8 c) {
    return __builtin_amdgcn_wmma_f32_16x16x32_bf16(false, a, false, b, (short)0, c, false, false);
}

// Fragment load for 16-bit A/B operands stored K-minor (row-major along K).
// ISA 16-bit A 16x32 layout: lane L -> row (L&15); lanes 0-15 hold K=0..7 &
// 16..23, lanes 16-31 hold K=8..15 & 24..31. Two 16B loads per lane.
template <typename V16, typename V8, typename E>
DEV V16 load_frag(const E* __restrict__ tile, int ld) {
    const int lane = (int)(threadIdx.x & 31u);
    const int row  = lane & 15;
    const int hs   = lane >> 4;
    const E* p = tile + (size_t)row * (size_t)ld + hs * 8;
    V8 lo = *reinterpret_cast<const V8*>(p);
    V8 hi = *reinterpret_cast<const V8*>(p + 16);
    return __builtin_shufflevector(lo, hi, 0, 1, 2, 3, 4, 5, 6, 7,
                                   8, 9, 10, 11, 12, 13, 14, 15);
}

// 16x16 f32 C tile store: lane L holds col (L&15); VGPR r holds row r+8*(L>>4)
DEV void store_tile_f32(float* __restrict__ C, int ldc, const f8& c) {
    const int lane = (int)(threadIdx.x & 31u);
    const int col  = lane & 15;
    const int hs   = lane >> 4;
#pragma unroll
    for (int r = 0; r < 8; ++r) C[(size_t)(r + 8 * hs) * ldc + col] = c[r];
}
DEV void store_tile_bf16(u16s* __restrict__ C, int ldc, const f8& c) {
    const int lane = (int)(threadIdx.x & 31u);
    const int col  = lane & 15;
    const int hs   = lane >> 4;
#pragma unroll
    for (int r = 0; r < 8; ++r)
        C[(size_t)(r + 8 * hs) * ldc + col] = f32_to_bf16(c[r]);
}

// ---------------------------------------------------------------------------
// K1/K9: LayerNorm (one wave32 per 512-wide row)
// ---------------------------------------------------------------------------
DEV void ln_row_stats(const float4 v[4], float& mean, float& rstd) {
    float s = 0.f, q = 0.f;
#pragma unroll
    for (int j = 0; j < 4; ++j) {
        s += v[j].x + v[j].y + v[j].z + v[j].w;
        q += v[j].x * v[j].x + v[j].y * v[j].y + v[j].z * v[j].z + v[j].w * v[j].w;
    }
#pragma unroll
    for (int o = 16; o > 0; o >>= 1) {
        s += __shfl_xor(s, o, 32);
        q += __shfl_xor(q, o, 32);
    }
    mean = s * (1.0f / DIM);
    float var = q * (1.0f / DIM) - mean * mean;
    rstd = rsqrtf(var + 1e-5f);
}

__global__ __launch_bounds__(256) void ln_to_f16_kernel(const float* __restrict__ X,
                                                        const float* __restrict__ gamma,
                                                        _Float16* __restrict__ Y) {
    const int row = blockIdx.x * 8 + (int)(threadIdx.x >> 5);
    const int l   = (int)(threadIdx.x & 31u);
    const float* xr = X + (size_t)row * DIM;
    float4 v[4];
#pragma unroll
    for (int j = 0; j < 4; ++j) v[j] = *reinterpret_cast<const float4*>(xr + 4 * l + 128 * j);
    float mean, rstd;
    ln_row_stats(v, mean, rstd);
    _Float16* yr = Y + (size_t)row * DIM;
#pragma unroll
    for (int j = 0; j < 4; ++j) {
        const int c = 4 * l + 128 * j;
        float4 g = *reinterpret_cast<const float4*>(gamma + c);
        yr[c + 0] = (_Float16)((v[j].x - mean) * rstd * g.x);
        yr[c + 1] = (_Float16)((v[j].y - mean) * rstd * g.y);
        yr[c + 2] = (_Float16)((v[j].z - mean) * rstd * g.z);
        yr[c + 3] = (_Float16)((v[j].w - mean) * rstd * g.w);
    }
}

__global__ __launch_bounds__(256) void ln_inplace_kernel(float* __restrict__ X,
                                                         const float* __restrict__ gamma) {
    const int row = blockIdx.x * 8 + (int)(threadIdx.x >> 5);
    const int l   = (int)(threadIdx.x & 31u);
    float* xr = X + (size_t)row * DIM;
    float4 v[4];
#pragma unroll
    for (int j = 0; j < 4; ++j) v[j] = *reinterpret_cast<const float4*>(xr + 4 * l + 128 * j);
    float mean, rstd;
    ln_row_stats(v, mean, rstd);
#pragma unroll
    for (int j = 0; j < 4; ++j) {
        const int c = 4 * l + 128 * j;
        float4 g = *reinterpret_cast<const float4*>(gamma + c);
        float4 o;
        o.x = (v[j].x - mean) * rstd * g.x;
        o.y = (v[j].y - mean) * rstd * g.y;
        o.z = (v[j].z - mean) * rstd * g.z;
        o.w = (v[j].w - mean) * rstd * g.w;
        *reinterpret_cast<float4*>(xr + c) = o;
    }
}

// ---------------------------------------------------------------------------
// K2: weight conversion  w_qkv -> f16, w_out -> bf16
// ---------------------------------------------------------------------------
__global__ __launch_bounds__(256) void wconv_kernel(const float* __restrict__ wq,
                                                    const float* __restrict__ wo,
                                                    _Float16* __restrict__ wq_h,
                                                    u16s* __restrict__ wo_u) {
    const int i = blockIdx.x * 256 + (int)threadIdx.x;
    if (i < QKVF * DIM) wq_h[i] = (_Float16)wq[i];
    if (i < DIM * DIM)  wo_u[i] = f32_to_bf16(wo[i]);
}

// ---------------------------------------------------------------------------
// K3/K8: generic NT GEMM, 128x128 block tile, 8 waves of 64x32, f32 output
//   C[m,n] = sum_k A[m,k] * B[n,k]      (A: MxK K-minor, B: NxK K-minor)
// ---------------------------------------------------------------------------
template <typename E, typename V16, typename V8>
__global__ __launch_bounds__(256) void gemm_nt_f32_kernel(const E* __restrict__ A,
                                                          const E* __restrict__ B,
                                                          float* __restrict__ C,
                                                          int N, int K) {
    const int w  = (int)(threadIdx.x >> 5);
    const int wm = w >> 2;          // 0..1
    const int wn = w & 3;           // 0..3
    const size_t m0 = (size_t)blockIdx.x * 128 + (size_t)wm * 64;
    const size_t n0 = (size_t)blockIdx.y * 128 + (size_t)wn * 32;

    f8 acc[4][2];
#pragma unroll
    for (int i = 0; i < 4; ++i)
#pragma unroll
        for (int j = 0; j < 2; ++j) acc[i][j] = f8zero();

    for (int k0 = 0; k0 < K; k0 += 32) {
        V16 af[4], bf2[2];
#pragma unroll
        for (int i = 0; i < 4; ++i)
            af[i] = load_frag<V16, V8>(A + (m0 + 16 * i) * (size_t)K + k0, K);
#pragma unroll
        for (int j = 0; j < 2; ++j)
            bf2[j] = load_frag<V16, V8>(B + (n0 + 16 * j) * (size_t)K + k0, K);
        if (k0 + 32 < K) {   // prefetch next K-slice (global_prefetch_b8)
            __builtin_prefetch(A + m0 * (size_t)K + k0 + 32, 0, 1);
            __builtin_prefetch(B + n0 * (size_t)K + k0 + 32, 0, 1);
        }
#pragma unroll
        for (int i = 0; i < 4; ++i)
#pragma unroll
            for (int j = 0; j < 2; ++j)
                acc[i][j] = wmma32(af[i], bf2[j], acc[i][j]);
    }
#pragma unroll
    for (int i = 0; i < 4; ++i)
#pragma unroll
        for (int j = 0; j < 2; ++j)
            store_tile_f32(C + (m0 + 16 * i) * (size_t)N + (n0 + 16 * j), N, acc[i][j]);
}

// ---------------------------------------------------------------------------
// K4: q feature-softmax (*scale) -> bf16 [bh, n, 64]; v -> bf16 transposed [bh, 64, n]
//   one wave per (row, head)
// ---------------------------------------------------------------------------
__global__ __launch_bounds__(256) void qv_prep_kernel(const float* __restrict__ qkv,
                                                      u16s* __restrict__ q_u,
                                                      u16s* __restrict__ vt_u) {
    const int bn = blockIdx.x;                   // 0..65535
    const int h  = (int)(threadIdx.x >> 5);      // wave = head
    const int l  = (int)(threadIdx.x & 31u);
    const int b  = bn >> 14;
    const int n  = bn & (SEQ - 1);
    const size_t base = (size_t)bn * QKVF + (size_t)h * DH;

    // q softmax over the 64-wide feature dim
    float x0 = qkv[base + l], x1 = qkv[base + l + 32];
    float m = fmaxf(x0, x1);
#pragma unroll
    for (int o = 16; o > 0; o >>= 1) m = fmaxf(m, __shfl_xor(m, o, 32));
    float e0 = __expf(x0 - m), e1 = __expf(x1 - m);
    float s = e0 + e1;
#pragma unroll
    for (int o = 16; o > 0; o >>= 1) s += __shfl_xor(s, o, 32);
    const float r = 0.125f / s;                  // scale = 64^-0.5 = 0.125
    const size_t bh = (size_t)b * HEADS + h;
    u16s* qd = q_u + (bh * SEQ + n) * DH;
    qd[l]      = f32_to_bf16(e0 * r);
    qd[l + 32] = f32_to_bf16(e1 * r);

    // v copy, transposed to [bh, e, n]
    float v0 = qkv[base + 2 * DIM + l], v1 = qkv[base + 2 * DIM + l + 32];
    vt_u[(bh * DH + l) * (size_t)SEQ + n]        = f32_to_bf16(v0);
    vt_u[(bh * DH + l + 32) * (size_t)SEQ + n]   = f32_to_bf16(v1);
}

// ---------------------------------------------------------------------------
// K5a: per-column (b,h,d) max and sum(exp) over n for k
// ---------------------------------------------------------------------------
__global__ __launch_bounds__(256) void k_stats_kernel(const float* __restrict__ qkv,
                                                      float* __restrict__ cmax,
                                                      float* __restrict__ csum) {
    __shared__ float red[256];
    const int col = blockIdx.x;                  // bh*64 + d,  0..2047
    const int bh  = col >> 6, d = col & 63;
    const int b   = bh >> 3,  h = bh & 7;
    const size_t coff = (size_t)DIM + (size_t)h * DH + d;
    const int tid = (int)threadIdx.x;

    float vals[64];
    float m = -3.4e38f;
#pragma unroll
    for (int i = 0; i < 64; ++i) {
        const int n = tid + 256 * i;
        float v = qkv[((size_t)b * SEQ + n) * QKVF + coff];
        vals[i] = v;
        m = fmaxf(m, v);
    }
    red[tid] = m;
    __syncthreads();
    for (int s = 128; s > 0; s >>= 1) {
        if (tid < s) red[tid] = fmaxf(red[tid], red[tid + s]);
        __syncthreads();
    }
    m = red[0];
    __syncthreads();
    float sum = 0.f;
#pragma unroll
    for (int i = 0; i < 64; ++i) sum += __expf(vals[i] - m);
    red[tid] = sum;
    __syncthreads();
    for (int s = 128; s > 0; s >>= 1) {
        if (tid < s) red[tid] += red[tid + s];
        __syncthreads();
    }
    if (tid == 0) { cmax[col] = m; csum[col] = red[0]; }
}

// K5b: normalize k -> bf16 transposed [bh, d, n]
__global__ __launch_bounds__(256) void k_norm_kernel(const float* __restrict__ qkv,
                                                     const float* __restrict__ cmax,
                                                     const float* __restrict__ csum,
                                                     u16s* __restrict__ kt_u) {
    const int bn = blockIdx.x;
    const int b  = bn >> 14;
    const int n  = bn & (SEQ - 1);
    for (int f = (int)threadIdx.x; f < DIM; f += 256) {
        const int h = f >> 6, d = f & 63;
        const int col = ((b * HEADS + h) << 6) + d;
        float x = qkv[(size_t)bn * QKVF + DIM + f];
        float r = __expf(x - cmax[col]) / csum[col];
        kt_u[(size_t)col * SEQ + n] = f32_to_bf16(r);
    }
}

// ---------------------------------------------------------------------------
// K6: context^T per (b,h): ctxT[e,d] = sum_n k[n,d]*v[n,e]
//   one block per bh; K=16384 split over 8 waves; LDS ds_add_f32 reduction
// ---------------------------------------------------------------------------
__global__ __launch_bounds__(256) void context_gemm_kernel(const u16s* __restrict__ kt_u,
                                                           const u16s* __restrict__ vt_u,
                                                           u16s* __restrict__ ctx_u) {
    __shared__ float cbuf[DH * DH];
    const int bh = blockIdx.x;
    const int w  = (int)(threadIdx.x >> 5);
    const __bf16* A = reinterpret_cast<const __bf16*>(kt_u) + (size_t)bh * DH * SEQ;
    const __bf16* B = reinterpret_cast<const __bf16*>(vt_u) + (size_t)bh * DH * SEQ;

    for (int idx = (int)threadIdx.x; idx < DH * DH; idx += 256) cbuf[idx] = 0.f;
    __syncthreads();

    f8 acc[4][4];
#pragma unroll
    for (int i = 0; i < 4; ++i)
#pragma unroll
        for (int j = 0; j < 4; ++j) acc[i][j] = f8zero();

    const int kbeg = w * (SEQ / 8);
    const int kend = kbeg + (SEQ / 8);
    for (int k0 = kbeg; k0 < kend; k0 += 32) {
        y16 af[4], bf4[4];
#pragma unroll
        for (int i = 0; i < 4; ++i)
            af[i] = load_frag<y16, y8>(A + (size_t)(16 * i) * SEQ + k0, SEQ);
#pragma unroll
        for (int j = 0; j < 4; ++j)
            bf4[j] = load_frag<y16, y8>(B + (size_t)(16 * j) * SEQ + k0, SEQ);
#pragma unroll
        for (int i = 0; i < 4; ++i)
#pragma unroll
            for (int j = 0; j < 4; ++j)
                acc[i][j] = wmma32(af[i], bf4[j], acc[i][j]);
    }

    const int lane = (int)(threadIdx.x & 31u);
    const int col  = lane & 15;
    const int hs   = lane >> 4;
#pragma unroll
    for (int i = 0; i < 4; ++i)
#pragma unroll
        for (int j = 0; j < 4; ++j)
#pragma unroll
            for (int r = 0; r < 8; ++r)
                atomicAdd(&cbuf[(16 * i + r + 8 * hs) * DH + 16 * j + col], acc[i][j][r]);
    __syncthreads();

    // write context transposed: ctxT[e][d] = context[d][e]
    for (int idx = (int)threadIdx.x; idx < DH * DH; idx += 256) {
        const int e = idx >> 6, d = idx & 63;
        ctx_u[(size_t)bh * DH * DH + idx] = f32_to_bf16(cbuf[d * DH + e]);
    }
}

// ---------------------------------------------------------------------------
// K7: out[n,e] = sum_d q[n,d] * ctxT[e,d] -> bf16 [row, h*64+e]
//   grid (SEQ/128, BH); 8 waves as 4(M) x 2(N), wave tile 32x32
// ---------------------------------------------------------------------------
__global__ __launch_bounds__(256) void attn_out_kernel(const u16s* __restrict__ q_u,
                                                       const u16s* __restrict__ ctx_u,
                                                       u16s* __restrict__ out_u) {
    const int bh = blockIdx.y;
    const int b  = bh >> 3, h = bh & 7;
    const int w  = (int)(threadIdx.x >> 5);
    const int wm = w >> 1;          // 0..3
    const int wn = w & 1;           // 0..1
    const __bf16* A = reinterpret_cast<const __bf16*>(q_u) + (size_t)bh * SEQ * DH;
    const __bf16* B = reinterpret_cast<const __bf16*>(ctx_u) + (size_t)bh * DH * DH;
    const size_t m0 = (size_t)blockIdx.x * 128 + (size_t)wm * 32;
    const int n0 = wn * 32;

    f8 acc[2][2];
#pragma unroll
    for (int i = 0; i < 2; ++i)
#pragma unroll
        for (int j = 0; j < 2; ++j) acc[i][j] = f8zero();

#pragma unroll
    for (int k0 = 0; k0 < DH; k0 += 32) {
        y16 af[2], bf2[2];
#pragma unroll
        for (int i = 0; i < 2; ++i)
            af[i] = load_frag<y16, y8>(A + (m0 + 16 * i) * (size_t)DH + k0, DH);
#pragma unroll
        for (int j = 0; j < 2; ++j)
            bf2[j] = load_frag<y16, y8>(B + (size_t)(n0 + 16 * j) * DH + k0, DH);
#pragma unroll
        for (int i = 0; i < 2; ++i)
#pragma unroll
            for (int j = 0; j < 2; ++j)
                acc[i][j] = wmma32(af[i], bf2[j], acc[i][j]);
    }

    u16s* Cb = out_u + ((size_t)b * SEQ + (size_t)blockIdx.x * 128) * DIM + h * DH;
#pragma unroll
    for (int i = 0; i < 2; ++i)
#pragma unroll
        for (int j = 0; j < 2; ++j)
            store_tile_bf16(Cb + (size_t)(wm * 32 + 16 * i) * DIM + n0 + 16 * j, DIM,
                            acc[i][j]);
}

// ---------------------------------------------------------------------------
// Launch
// ---------------------------------------------------------------------------
extern "C" void kernel_launch(void* const* d_in, const int* in_sizes, int n_in,
                              void* d_out, int out_size, void* d_ws, size_t ws_size,
                              hipStream_t stream) {
    const float* x        = (const float*)d_in[0];
    const float* ng       = (const float*)d_in[1];
    const float* w_qkv    = (const float*)d_in[2];
    const float* w_out    = (const float*)d_in[3];
    const float* og       = (const float*)d_in[4];
    float*       out      = (float*)d_out;

    char* ws = (char*)d_ws;
    size_t off = 0;
    auto alloc = [&](size_t bytes) { char* p = ws + off; off += (bytes + 255) & ~(size_t)255; return p; };

    _Float16* xn_h   = (_Float16*)alloc((size_t)ROWS * DIM * 2);   // 64 MiB
    _Float16* wqkv_h = (_Float16*)alloc((size_t)QKVF * DIM * 2);   // 1.5 MiB
    u16s*     wout_u = (u16s*)    alloc((size_t)DIM * DIM * 2);    // 0.5 MiB
    float*    qkv    = (float*)   alloc((size_t)ROWS * QKVF * 4);  // 384 MiB
    u16s*     q_u    = (u16s*)    alloc((size_t)ROWS * DIM * 2);   // 64 MiB
    u16s*     kt_u   = (u16s*)    alloc((size_t)ROWS * DIM * 2);   // 64 MiB
    u16s*     vt_u   = (u16s*)    alloc((size_t)ROWS * DIM * 2);   // 64 MiB
    u16s*     ctx_u  = (u16s*)    alloc((size_t)BH * DH * DH * 2); // 256 KiB
    u16s*     out_u  = (u16s*)    alloc((size_t)ROWS * DIM * 2);   // 64 MiB
    float*    cmax   = (float*)   alloc((size_t)BH * DH * 4);
    float*    csum   = (float*)   alloc((size_t)BH * DH * 4);
    (void)ws_size; (void)in_sizes; (void)n_in; (void)out_size;

    // 1) LayerNorm(x) -> f16
    ln_to_f16_kernel<<<ROWS / 8, 256, 0, stream>>>(x, ng, xn_h);

    // 2) weight conversions
    wconv_kernel<<<(QKVF * DIM + 255) / 256, 256, 0, stream>>>(w_qkv, w_out, wqkv_h, wout_u);

    // 3) qkv = xn * w_qkv^T   (f16 WMMA, M=65536 N=1536 K=512)
    gemm_nt_f32_kernel<_Float16, h16, h8>
        <<<dim3(ROWS / 128, QKVF / 128), 256, 0, stream>>>(xn_h, wqkv_h, qkv, QKVF, DIM);

    // 4) q softmax(+scale) and v transpose
    qv_prep_kernel<<<ROWS, 256, 0, stream>>>(qkv, q_u, vt_u);

    // 5) k sequence-softmax (column stats, then normalize+transpose)
    k_stats_kernel<<<BH * DH, 256, 0, stream>>>(qkv, cmax, csum);
    k_norm_kernel<<<ROWS, 256, 0, stream>>>(qkv, cmax, csum, kt_u);

    // 6) context^T = (k^T v)^T per head  (bf16 WMMA, K=16384, LDS reduce)
    context_gemm_kernel<<<BH, 256, 0, stream>>>(kt_u, vt_u, ctx_u);

    // 7) out = q * context  (bf16 WMMA) -> bf16 [row, hidden]
    attn_out_kernel<<<dim3(SEQ / 128, BH), 256, 0, stream>>>(q_u, ctx_u, out_u);

    // 8) projection: out * w_out^T  (bf16 WMMA, M=65536 N=512 K=512) -> f32 d_out
    gemm_nt_f32_kernel<__bf16, y16, y8>
        <<<dim3(ROWS / 128, DIM / 128), 256, 0, stream>>>(
            reinterpret_cast<const __bf16*>(out_u),
            reinterpret_cast<const __bf16*>(wout_u), out, DIM, DIM);

    // 9) final LayerNorm in place on d_out
    ln_inplace_kernel<<<ROWS / 8, 256, 0, stream>>>(out, og);
}
